// GraphCNN_4191888081351
// MI455X (gfx1250) — matrix operand order repaired and checked by reading, not compiled, not run on hardware
//
#include <hip/hip_runtime.h>

typedef __attribute__((ext_vector_type(16))) __bf16 v16bf;
typedef __attribute__((ext_vector_type(8)))  float  v8f;

__device__ __forceinline__ unsigned short f2bf(float f) {
  unsigned u = __float_as_uint(f);
  u += 0x7FFFu + ((u >> 16) & 1u);          // round-to-nearest-even
  return (unsigned short)(u >> 16);
}

// ---------------- elementwise helpers ----------------
__global__ void k_zero(float* __restrict__ p, long n) {
  long i = blockIdx.x * (long)blockDim.x + threadIdx.x;
  if (i < n) p[i] = 0.0f;
}

__global__ void k_cvt(const float* __restrict__ x, unsigned short* __restrict__ y, long n) {
  long i = blockIdx.x * (long)blockDim.x + threadIdx.x;
  if (i < n) y[i] = f2bf(x[i]);
}

// out_bf = bf16( pooled + (1+eps[l]) * h )
__global__ void k_eps_cvt(const float* __restrict__ pooled, const float* __restrict__ h,
                          const float* __restrict__ eps, int l,
                          unsigned short* __restrict__ out, long n) {
  long i = blockIdx.x * (long)blockDim.x + threadIdx.x;
  if (i >= n) return;
  float e = 1.0f + eps[l];
  out[i] = f2bf(pooled[i] + e * h[i]);
}

// ---------------- graph scatter ----------------
__global__ void k_scatter(const float* __restrict__ h, const int* __restrict__ edges,
                          int E, int fshift, float* __restrict__ pooled) {
  long idx = blockIdx.x * (long)blockDim.x + threadIdx.x;
  long total = (long)E << fshift;           // one thread = 4 floats
  if (idx >= total) return;
  long e  = idx >> fshift;
  int  f4 = (int)(idx - (e << fshift));
  int  F  = 4 << fshift;
  int dst = edges[e];
  int src = edges[(long)E + e];
  const float* hp = h + (size_t)src * F + 4 * f4;
  float* pp = pooled + (size_t)dst * F + 4 * f4;
  atomicAdd(pp + 0, hp[0]);
  atomicAdd(pp + 1, hp[1]);
  atomicAdd(pp + 2, hp[2]);
  atomicAdd(pp + 3, hp[3]);
}

__global__ void k_seg_pool(const float* __restrict__ h, const int* __restrict__ gid,
                           int N, int fshift, float* __restrict__ out) {
  long idx = blockIdx.x * (long)blockDim.x + threadIdx.x;
  long total = (long)N << fshift;
  if (idx >= total) return;
  long i  = idx >> fshift;
  int  f4 = (int)(idx - (i << fshift));
  int  F  = 4 << fshift;
  int  g  = gid[i];
  const float* hp = h + (size_t)i * F + 4 * f4;
  float* pp = out + (size_t)g * F + 4 * f4;
  atomicAdd(pp + 0, hp[0]);
  atomicAdd(pp + 1, hp[1]);
  atomicAdd(pp + 2, hp[2]);
  atomicAdd(pp + 3, hp[3]);
}

// ---------------- batch norm (feature dim fixed at 512) ----------------
__global__ void k_bn_stats(const float* __restrict__ x, float* __restrict__ sums) {
  int col = blockIdx.x * 256 + threadIdx.x;       // grid.x = 2
  int r0  = blockIdx.y * 512;
  float s1 = 0.f, s2 = 0.f;
  for (int r = 0; r < 512; ++r) {
    float v = x[(size_t)(r0 + r) * 512 + col];
    s1 += v; s2 += v * v;
  }
  atomicAdd(&sums[col], s1);
  atomicAdd(&sums[512 + col], s2);
}

__global__ void k_bn_relu_cvt(const float* __restrict__ x, const float* __restrict__ sums,
                              const float* __restrict__ gw, const float* __restrict__ bw,
                              int M, float* __restrict__ of32,
                              unsigned short* __restrict__ obf) {
  long idx = blockIdx.x * (long)blockDim.x + threadIdx.x;
  if (idx >= (long)M * 512) return;
  int col = (int)(idx & 511);
  float invM = 1.0f / (float)M;
  float m = sums[col] * invM;
  float v = sums[512 + col] * invM - m * m;
  float y = gw[col] * (x[idx] - m) * rsqrtf(v + 1e-5f) + bw[col];
  y = fmaxf(y, 0.0f);
  if (of32) of32[idx] = y;
  if (obf)  obf[idx]  = f2bf(y);
}

// ---------------- transpose + convert motif2A [R,C]f32 -> [C,R]bf16 ----------------
__global__ void k_transpose_cvt(const float* __restrict__ in, unsigned short* __restrict__ out,
                                int R, int C) {
  __shared__ float tile[32][33];
  int c0 = blockIdx.x * 32;
  int r0 = blockIdx.y * 32;
  int tx = threadIdx.x, ty = threadIdx.y;         // block (32,8)
  for (int j = 0; j < 32; j += 8)
    tile[ty + j][tx] = in[(size_t)(r0 + ty + j) * C + c0 + tx];
  __syncthreads();
  for (int j = 0; j < 32; j += 8)
    out[(size_t)(c0 + ty + j) * R + r0 + tx] = f2bf(tile[tx][ty + j]);
}

// ---------------- attention fuse + concat ----------------
__global__ void k_concat(const float* __restrict__ ha, const float* __restrict__ hb,
                         const float* __restrict__ atts, float* __restrict__ out, long n) {
  long idx = blockIdx.x * (long)blockDim.x + threadIdx.x;
  if (idx >= n) return;
  float e0 = __expf(atts[0]), e1 = __expf(atts[1]);
  float inv = 1.0f / (e0 + e1);
  float a0 = e0 * inv, a1 = e1 * inv;
  long i = idx >> 9;
  int  j = (int)(idx & 511);
  out[i * 1024 + j]       = a0 * ha[idx];
  out[i * 1024 + 512 + j] = a1 * hb[idx];
}

// ---------------- WMMA bf16 GEMM: C[M,N]f32 = A[M,K]bf16 @ B[K,N]bf16 (+bias) ----------
// block 256 = 8 waves; tile 128x64, BK=32; wave -> 16x64 strip (4 wmma tiles)
#define PA 48
#define PB 48
__global__ __launch_bounds__(256) void k_gemm_bf16(
    const unsigned short* __restrict__ A, const unsigned short* __restrict__ B,
    float* __restrict__ C, const float* __restrict__ bias, int M, int N, int K) {
  __shared__ unsigned short sA[128 * PA];   // [row][k], fragments 16B-aligned
  __shared__ unsigned short sBt[64 * PB];   // [n][k]
  int tid = threadIdx.x;
  int wave = tid >> 5, lane = tid & 31;
  int m0 = blockIdx.y * 128;
  int n0 = blockIdx.x * 64;

  v8f acc[4];
#pragma unroll
  for (int t = 0; t < 4; ++t) acc[t] = (v8f){0, 0, 0, 0, 0, 0, 0, 0};

  int row = lane & 15, hi = lane >> 4;

  for (int k0 = 0; k0 < K; k0 += 32) {
    // stage A: 128x32 bf16, 2 x uint4 per thread
    int chunk = tid * 2;
#pragma unroll
    for (int c = 0; c < 2; ++c, ++chunk) {
      int r  = chunk >> 2;
      int kk = (chunk & 3) << 3;
      *(uint4*)(sA + r * PA + kk) =
          *(const uint4*)(A + (size_t)(m0 + r) * K + k0 + kk);
    }
    // stage B transposed: 32x64 bf16, one uint4 per thread scattered by column
    {
      int kr = tid >> 3;
      int nn = (tid & 7) << 3;
      uint4 v = *(const uint4*)(B + (size_t)(k0 + kr) * N + n0 + nn);
      const unsigned short* e = (const unsigned short*)&v;
#pragma unroll
      for (int j = 0; j < 8; ++j) sBt[(nn + j) * PB + kr] = e[j];
    }
    __syncthreads();

    union Frag { uint4 q[2]; v16bf v; };
    // A fragment (16-bit A 16x32 layout): halves 0-7 -> K = hi*8..hi*8+7,
    // halves 8-15 -> K = 16+hi*8..16+hi*8+7
    Frag af;
    const unsigned short* abase = sA + (wave * 16 + row) * PA;
    af.q[0] = *(const uint4*)(abase + hi * 8);
    af.q[1] = *(const uint4*)(abase + 16 + hi * 8);
    // B fragments (16-bit B 32x16 layout): lane holds column, same K as A:
    // halves 0-7 -> K = hi*8.., halves 8-15 -> K = 16+hi*8...  Preload all 4 tiles so the LDS
    // latency overlaps the WMMA chain instead of a wait per tile.
    Frag bfr[4];
#pragma unroll
    for (int t = 0; t < 4; ++t) {
      const unsigned short* bbase = sBt + (t * 16 + row) * PB + hi * 8;
      bfr[t].q[0] = *(const uint4*)(bbase);
      bfr[t].q[1] = *(const uint4*)(bbase + 16);
    }
#pragma unroll
    for (int t = 0; t < 4; ++t) {
      acc[t] = __builtin_amdgcn_wmma_f32_16x16x32_bf16(
          false, af.v, false, bfr[t].v, (short)0, acc[t], false, false);
    }
    __syncthreads();
  }

#pragma unroll
  for (int t = 0; t < 4; ++t) {
    int col = n0 + t * 16 + row;
    float bv = bias ? bias[col] : 0.0f;
#pragma unroll
    for (int v = 0; v < 8; ++v) {
      int r = m0 + wave * 16 + hi * 8 + v;
      C[(size_t)r * N + col] = acc[t][v] + bv;
    }
  }
}

// ---------------- prediction heads ----------------
__global__ void k_head_init(const float* __restrict__ b0, const float* __restrict__ bl,
                            float* __restrict__ out) {
  int t = threadIdx.x;                // 640
  int o = t % 10;
  out[t] = b0[o] + bl[o] + bl[10 + o] + bl[20 + o] + bl[30 + o];
}

__global__ void k_head(const float* __restrict__ pooled, const float* __restrict__ W,
                       int F, float* __restrict__ out) {
  int t = threadIdx.x;                // 640
  int g = t / 10, o = t % 10;
  float s = 0.f;
  for (int f = 0; f < F; ++f) s += pooled[(size_t)g * F + f] * W[f * 10 + o];
  out[t] += s;
}

// ---------------- host orchestration ----------------
static inline unsigned gdiv(long n, int b) { return (unsigned)((n + b - 1) / b); }

extern "C" void kernel_launch(void* const* d_in, const int* in_sizes, int n_in,
                              void* d_out, int out_size, void* d_ws, size_t ws_size,
                              hipStream_t stream) {
  (void)in_sizes; (void)n_in; (void)out_size; (void)ws_size;
  const int N1 = 8192, N2 = 4096, E1 = 262144, E2 = 65536, H = 512, G = 64, L = 4;

  const float* x1    = (const float*)d_in[0];
  const float* x2    = (const float*)d_in[1];
  const float* motif = (const float*)d_in[2];
  const int*   edge1 = (const int*)d_in[3];
  const int*   edge2 = (const int*)d_in[4];
  const int*   gids  = (const int*)d_in[5];
  const float* eps   = (const float*)d_in[6];
  const float* atts  = (const float*)d_in[7];
  const float* m1W10 = (const float*)d_in[8];
  const float* m1W1r = (const float*)d_in[9];
  const float* m1b1  = (const float*)d_in[10];
  const float* m1g   = (const float*)d_in[11];
  const float* m1bb  = (const float*)d_in[12];
  const float* m1W2  = (const float*)d_in[13];
  const float* m1b2  = (const float*)d_in[14];
  const float* bn1g  = (const float*)d_in[15];
  const float* bn1b  = (const float*)d_in[16];
  const float* m2W10 = (const float*)d_in[17];
  const float* m2W1r = (const float*)d_in[18];
  const float* m2b1  = (const float*)d_in[19];
  const float* m2g   = (const float*)d_in[20];
  const float* m2bb  = (const float*)d_in[21];
  const float* m2W2  = (const float*)d_in[22];
  const float* m2b2  = (const float*)d_in[23];
  const float* bn2g  = (const float*)d_in[24];
  const float* bn2b  = (const float*)d_in[25];
  const float* pW0   = (const float*)d_in[26];
  const float* pb0   = (const float*)d_in[27];
  const float* pW    = (const float*)d_in[28];
  const float* pb    = (const float*)d_in[29];
  float* out = (float*)d_out;

  char* w = (char*)d_ws;
  size_t off = 0;
  auto take = [&](size_t bytes) -> void* {
    void* p = w + off;
    off = (off + bytes + 255) & ~(size_t)255;
    return p;
  };
  unsigned short* motifT  = (unsigned short*)take((size_t)N1 * N2 * 2);   // 64 MB -> L2-resident
  float*          h1buf   = (float*)take((size_t)N1 * 1024 * 4);
  float*          h2buf   = (float*)take((size_t)N2 * 512 * 4);
  float*          pooled1 = (float*)take((size_t)N1 * 1024 * 4);
  float*          pooled2 = (float*)take((size_t)N2 * 512 * 4);
  unsigned short* A1bf    = (unsigned short*)take((size_t)N1 * 1024 * 2);
  unsigned short* A2bf    = (unsigned short*)take((size_t)N2 * 512 * 2);
  float*          rep1    = (float*)take((size_t)N1 * 512 * 4);
  float*          rep2    = (float*)take((size_t)N2 * 512 * 4);
  unsigned short* t1bf    = (unsigned short*)take((size_t)N1 * 512 * 2);
  unsigned short* t2bf    = (unsigned short*)take((size_t)N2 * 512 * 2);
  unsigned short* h2bf    = (unsigned short*)take((size_t)N2 * 512 * 2);
  unsigned short* W1bf    = (unsigned short*)take((size_t)1024 * 512 * 2);
  unsigned short* W2bf    = (unsigned short*)take((size_t)512 * 512 * 2);
  float*          bns     = (float*)take(1024 * 4);
  float*          gpool   = (float*)take((size_t)G * 1024 * 4);
  float*          h1br    = pooled1;                                  // dead after eps_cvt -> reuse
  float*          h2A     = (float*)((char*)pooled1 + (size_t)N1 * 512 * 4);

  auto zero = [&](float* p, long n) { k_zero<<<gdiv(n, 256), 256, 0, stream>>>(p, n); };
  auto cvt  = [&](const float* x, unsigned short* y, long n) {
    k_cvt<<<gdiv(n, 256), 256, 0, stream>>>(x, y, n);
  };
  auto gemm = [&](const unsigned short* A, const unsigned short* B, float* C,
                  const float* bias, int M, int N, int K) {
    dim3 g(N / 64, M / 128);
    k_gemm_bf16<<<g, 256, 0, stream>>>(A, B, C, bias, M, N, K);
  };
  auto bnstats = [&](const float* x, int M) {
    zero(bns, 1024);
    k_bn_stats<<<dim3(2, M / 512), 256, 0, stream>>>(x, bns);
  };

  // one-time: motif2A^T -> bf16
  k_transpose_cvt<<<dim3(N1 / 32, N2 / 32), dim3(32, 8), 0, stream>>>(motif, motifT, N2, N1);

  // readout init + hidden[0] head
  k_head_init<<<1, 640, 0, stream>>>(pb0, pb, out);
  zero(gpool, (long)G * 128);
  k_seg_pool<<<gdiv((long)N1 << 5, 256), 256, 0, stream>>>(x1, gids, N1, 5, gpool);
  k_head<<<1, 640, 0, stream>>>(gpool, pW0, 128, out);

  const float* h1 = x1; int F1 = 128;
  const float* h2 = x2; int F2 = 64;

  for (int l = 0; l < L; ++l) {
    int fs1 = (F1 == 128) ? 5 : 8;
    int fs2 = (F2 == 64) ? 4 : 7;

    // ---- base-graph branch ----
    zero(pooled1, (long)N1 * F1);
    k_scatter<<<gdiv((long)E1 << fs1, 256), 256, 0, stream>>>(h1, edge1, E1, fs1, pooled1);
    k_eps_cvt<<<gdiv((long)N1 * F1, 256), 256, 0, stream>>>(pooled1, h1, eps, l, A1bf, (long)N1 * F1);
    const float* W1sel = (l == 0) ? m1W10 : m1W1r + (size_t)(l - 1) * 1024 * 512;
    cvt(W1sel, W1bf, (long)F1 * 512);
    gemm(A1bf, W1bf, rep1, m1b1 + l * H, N1, 512, F1);
    bnstats(rep1, N1);
    k_bn_relu_cvt<<<gdiv((long)N1 * 512, 256), 256, 0, stream>>>(rep1, bns, m1g + l * H, m1bb + l * H, N1, nullptr, t1bf);
    cvt(m1W2 + (size_t)l * 512 * 512, W2bf, 512 * 512);
    gemm(t1bf, W2bf, rep1, m1b2 + l * H, N1, 512, 512);
    bnstats(rep1, N1);
    k_bn_relu_cvt<<<gdiv((long)N1 * 512, 256), 256, 0, stream>>>(rep1, bns, bn1g + l * H, bn1b + l * H, N1, h1br, nullptr);

    // ---- hyper-graph branch ----
    zero(pooled2, (long)N2 * F2);
    k_scatter<<<gdiv((long)E2 << fs2, 256), 256, 0, stream>>>(h2, edge2, E2, fs2, pooled2);
    k_eps_cvt<<<gdiv((long)N2 * F2, 256), 256, 0, stream>>>(pooled2, h2, eps, l, A2bf, (long)N2 * F2);
    const float* W1hsel = (l == 0) ? m2W10 : m2W1r + (size_t)(l - 1) * 512 * 512;
    cvt(W1hsel, W1bf, (long)F2 * 512);
    gemm(A2bf, W1bf, rep2, m2b1 + l * H, N2, 512, F2);
    bnstats(rep2, N2);
    k_bn_relu_cvt<<<gdiv((long)N2 * 512, 256), 256, 0, stream>>>(rep2, bns, m2g + l * H, m2bb + l * H, N2, nullptr, t2bf);
    cvt(m2W2 + (size_t)l * 512 * 512, W2bf, 512 * 512);
    gemm(t2bf, W2bf, rep2, m2b2 + l * H, N2, 512, 512);
    bnstats(rep2, N2);
    k_bn_relu_cvt<<<gdiv((long)N2 * 512, 256), 256, 0, stream>>>(rep2, bns, bn2g + l * H, bn2b + l * H, N2, h2buf, h2bf);

    // ---- dominant GEMM: h2A = motif2A^T @ h2  (8192x512x4096, A served from L2) ----
    gemm(motifT, h2bf, h2A, nullptr, N1, 512, N2);

    // ---- attention fuse + concat -> h1buf [N1,1024] ----
    k_concat<<<gdiv((long)N1 * 512, 256), 256, 0, stream>>>(h1br, h2A, atts, h1buf, (long)N1 * 512);

    // ---- jumping-knowledge head for hidden[l+1] ----
    zero(gpool, (long)G * 1024);
    k_seg_pool<<<gdiv((long)N1 << 8, 256), 256, 0, stream>>>(h1buf, gids, N1, 8, gpool);
    k_head<<<1, 640, 0, stream>>>(gpool, pW + (size_t)l * 1024 * 10, 1024, out);

    h1 = h1buf; F1 = 1024;
    h2 = h2buf; F2 = 512;
  }
}
